// SwinTransformer3D_84052509983043
// MI455X (gfx1250) — matrix operand order repaired and checked by reading, not compiled, not run on hardware
//
#include <hip/hip_runtime.h>
#include <hip/hip_bf16.h>

typedef __attribute__((ext_vector_type(16))) _Float16 v16h;
typedef __attribute__((ext_vector_type(8)))  float    v8f;

#define DEV static __device__ __forceinline__

// ---------------- problem dims ----------------
constexpr int C    = 96;        // embed dim
constexpr int NH   = 3;         // heads
constexpr int HD   = 32;        // head dim
constexpr int C3   = 288;       // 3*C
constexpr int HID  = 384;       // mlp hidden
constexpr int NTOK = 98;        // tokens per window (2*7*7)
constexpr int NPAD = 112;       // padded to 7 WMMA tiles
constexpr int NP2  = 128;       // q/k/v token pad
constexpr int NWIN = 2048;      // B(2) * nW(1024)
constexpr int NRPB = 507;       // (2*2-1)*13*13 rel-pos table rows
constexpr int TOKENS = 2 * 8 * 112 * 112; // 200704

// ---------------- workspace layout (bytes) ----------------
constexpr size_t OFF_QKVW = 0;
constexpr size_t OFF_PROJW = OFF_QKVW + (size_t)C3 * C * 2;
constexpr size_t OFF_FC1W  = OFF_PROJW + (size_t)C * C * 2;
constexpr size_t OFF_FC2W  = OFF_FC1W + (size_t)HID * C * 2;
constexpr size_t OFF_WINA  = OFF_FC2W + (size_t)C * HID * 2;   // [2048][112][96] f16, aliased as attnout later
constexpr size_t OFF_Q     = OFF_WINA + (size_t)NWIN * NPAD * C * 2;
constexpr size_t OFF_K     = OFF_Q + (size_t)NWIN * NH * NP2 * HD * 2;
constexpr size_t OFF_VT    = OFF_K + (size_t)NWIN * NH * NP2 * HD * 2;

// ---------------- dynamic LDS partitions ----------------
constexpr unsigned SM_ATTN_S   = 0;                       // float  [112*112] 50176
constexpr unsigned SM_ATTN_PV  = 50176;                   // f16    [112*128] 28672
constexpr unsigned SM_ATTN_KL  = 78848;                   // f16    [128*32]   8192
constexpr unsigned SM_ATTN_VL  = 87040;                   // f16    [32*128]   8192
constexpr unsigned SM_ATTN_RPB = 95232;                   // float  [507]      2028
constexpr unsigned SM_ATTN_CRG = 97264;                   // int    [112]       448
constexpr unsigned SM_ATTN_SZ  = 97728;

constexpr unsigned SM_MLP_A  = 0;                         // f16 [128*96]  24576
constexpr unsigned SM_MLP_H  = 24576;                     // f16 [128*384] 98304
constexpr unsigned SM_MLP_SZ = 24576 + 98304;             // 122880

// ---------------- WMMA helpers ----------------
DEV v8f WMMA(v16h a, v16h b, v8f c) {
  // v_wmma_f32_16x16x32_f16
  return __builtin_amdgcn_wmma_f32_16x16x32_f16(false, a, false, b, (short)0, c,
                                                false, false);
}

// A-matrix 16x32 f16 tile from row-major matrix (p -> tile origin, ld elems/row).
// Lane L<16 holds row M=L, K {0..7,16..23}; L>=16 holds row M=L-16, K {8..15,24..31}.
DEV v16h load_a16(const _Float16* p, int ld) {
  int lane = threadIdx.x & 31;
  const _Float16* row = p + (lane & 15) * ld + ((lane < 16) ? 0 : 8);
  union { v16h v; unsigned int u[8]; } r;
#pragma unroll
  for (int i = 0; i < 8; ++i) {
    int k = ((i < 4) ? 0 : 16) + 2 * (i & 3);
    r.u[i] = *(const unsigned int*)(row + k);
  }
  return r.v;
}

// B-matrix 32x16 f16 tile where B[k][n] = W[n][k], W row-major (p -> W[nbase][kbase]).
// Lane L<16: col n=L, K {0..7,16..23}; L>=16: col n=L-16, K {8..15,24..31}.
DEV v16h load_bw(const _Float16* p, int ld) {
  int lane = threadIdx.x & 31;
  const _Float16* row = p + (lane & 15) * ld + ((lane < 16) ? 0 : 8);
  union { v16h v; unsigned int u[8]; } r;
#pragma unroll
  for (int i = 0; i < 8; ++i) r.u[i] = *(const unsigned int*)(row + ((i < 4) ? 0 : 16) + 2 * (i & 3));
  return r.v;
}

// cooperative 16B-chunk copy (global <-> LDS), bytes must be multiple of 16
DEV void copy16(const void* src, void* dst, int bytes, int tid, int nthr) {
  const uint4* s = (const uint4*)src;
  uint4* d = (uint4*)dst;
  int n = bytes >> 4;
  for (int i = tid; i < n; i += nthr) d[i] = s[i];
}

DEV void win_decompose(int w, int& b_, int& wdi, int& whi, int& wwi) {
  b_ = w >> 10; int r = w & 1023;
  wdi = r >> 8; whi = (r >> 4) & 15; wwi = r & 15;
}
DEV void tok_decompose(int t, int& od, int& oh, int& ow) {
  od = t / 49; int r = t - od * 49; oh = r / 7; ow = r - oh * 7;
}

// ---------------- K0: weights fp32 -> f16 ----------------
__global__ void k_prep(const float* qkvw, const float* projw, const float* fc1w,
                       const float* fc2w, _Float16* o_qkvw, _Float16* o_projw,
                       _Float16* o_fc1w, _Float16* o_fc2w) {
  int i = blockIdx.x * blockDim.x + threadIdx.x;
  if (i < C3 * C)  o_qkvw[i]  = (_Float16)qkvw[i];
  if (i < C * C)   o_projw[i] = (_Float16)projw[i];
  if (i < HID * C) o_fc1w[i]  = (_Float16)fc1w[i];
  if (i < C * HID) o_fc2w[i]  = (_Float16)fc2w[i];
}

// ---------------- K1: LN1 + shifted roll + window partition -> f16 ----------------
__global__ void k_ln1(const float* __restrict__ x, const float* __restrict__ n1w,
                      const float* __restrict__ n1b, _Float16* __restrict__ winA) {
  int slot = blockIdx.x * 4 + (threadIdx.x >> 5);   // one wave per token slot
  int lane = threadIdx.x & 31;
  int w = slot / NPAD, m = slot % NPAD;
  _Float16* dst = winA + ((size_t)w * NPAD + m) * C;
  if (m >= NTOK) {  // zero pad rows so QKV GEMM sees clean A tiles
    dst[lane] = (_Float16)0.f; dst[lane + 32] = (_Float16)0.f; dst[lane + 64] = (_Float16)0.f;
    return;
  }
  int b_, wdi, whi, wwi; win_decompose(w, b_, wdi, whi, wwi);
  int od, oh, ow; tok_decompose(m, od, oh, ow);
  int sd = (wdi * 2 + od + 1) & 7;                       // roll(-1) on D=8
  int sh = whi * 7 + oh + 3; if (sh >= 112) sh -= 112;   // roll(-3) on H
  int sw = wwi * 7 + ow + 3; if (sw >= 112) sw -= 112;   // roll(-3) on W
  const float* src = x + ((((size_t)b_ * 8 + sd) * 112 + sh) * 112 + sw) * C;
  float f0 = src[lane], f1 = src[lane + 32], f2 = src[lane + 64];
  float s = f0 + f1 + f2;
#pragma unroll
  for (int off = 16; off; off >>= 1) s += __shfl_xor(s, off, 32);
  float mu = s * (1.f / 96.f);
  float d0 = f0 - mu, d1 = f1 - mu, d2 = f2 - mu;
  float q = d0 * d0 + d1 * d1 + d2 * d2;
#pragma unroll
  for (int off = 16; off; off >>= 1) q += __shfl_xor(q, off, 32);
  float rs = rsqrtf(q * (1.f / 96.f) + 1e-5f);
  dst[lane]      = (_Float16)(d0 * rs * n1w[lane] + n1b[lane]);
  dst[lane + 32] = (_Float16)(d1 * rs * n1w[lane + 32] + n1b[lane + 32]);
  dst[lane + 64] = (_Float16)(d2 * rs * n1w[lane + 64] + n1b[lane + 64]);
}

// ---------------- K2: QKV GEMM per window: A window staged in LDS, B held in VGPRs ----------------
__global__ void __launch_bounds__(128, 1)
k_qkv(const _Float16* __restrict__ winA, const _Float16* __restrict__ qkvw,
      const float* __restrict__ qkvb, _Float16* __restrict__ qb,
      _Float16* __restrict__ kb, _Float16* __restrict__ vT) {
  __shared__ _Float16 AL[NPAD * C];   // 21504 B
  int w = blockIdx.x;
  int tid = threadIdx.x, wave = tid >> 5, lane = tid & 31;
  const _Float16* Aw = winA + (size_t)w * NPAD * C;
  __builtin_prefetch(Aw, 0, 1);   // global_prefetch_b8
  copy16(Aw, AL, NPAD * C * 2, tid, 128);
  __syncthreads();

  for (int nt = wave; nt < 18; nt += 4) {
    // B tiles for this 16-wide output stripe: held in VGPRs across all 7 M tiles
    v16h b0 = load_bw(qkvw + nt * 16 * C + 0,  C);
    v16h b1 = load_bw(qkvw + nt * 16 * C + 32, C);
    v16h b2 = load_bw(qkvw + nt * 16 * C + 64, C);
    // three / head / q-scale are wave-uniform functions of nt (16-wide tiles)
    int three = nt / 6;
    int nm = nt - three * 6;
    int head = nm >> 1;
    int d = (nm & 1) * 16 + (lane & 15);
    float sc = (three == 0) ? 0.17677669529663689f : 1.0f;  // hd^-0.5 on q (after bias)
    float bia = qkvb[nt * 16 + (lane & 15)];
    int roff = (lane < 16) ? 0 : 8;
    size_t hb = (size_t)(w * NH + head);
    for (int mt = 0; mt < 7; ++mt) {
      v8f acc = {};
      acc = WMMA(load_a16(AL + mt * 16 * C + 0,  C), b0, acc);
      acc = WMMA(load_a16(AL + mt * 16 * C + 32, C), b1, acc);
      acc = WMMA(load_a16(AL + mt * 16 * C + 64, C), b2, acc);
      if (three == 0) {
        _Float16* dq = qb + hb * NP2 * HD + d;
#pragma unroll
        for (int r = 0; r < 8; ++r)
          dq[(mt * 16 + r + roff) * HD] = (_Float16)((acc[r] + bia) * sc);
      } else if (three == 1) {
        _Float16* dk = kb + hb * NP2 * HD + d;
#pragma unroll
        for (int r = 0; r < 8; ++r)
          dk[(mt * 16 + r + roff) * HD] = (_Float16)(acc[r] + bia);
      } else {
        _Float16* dv = vT + (hb * HD + d) * NP2;   // v stored transposed
#pragma unroll
        for (int r = 0; r < 8; ++r)
          dv[mt * 16 + r + roff] = (_Float16)(acc[r] + bia);
      }
    }
  }
  // zero token pad rows 112..127 (keeps WMMA operands finite)
  for (int i = tid; i < NH * 16 * HD; i += 128) {
    int head = i / (16 * HD); int r = i % (16 * HD);
    int m = NP2 - 16 + r / HD; int d = r % HD;
    size_t hb = (size_t)(w * NH + head);
    qb[(hb * NP2 + m) * HD + d] = (_Float16)0.f;
    kb[(hb * NP2 + m) * HD + d] = (_Float16)0.f;
    vT[(hb * HD + d) * NP2 + m] = (_Float16)0.f;
  }
}

// ---------------- K3: attention per (window, head): S=QK^T, softmax, P@V ----------------
__global__ void __launch_bounds__(128, 1)
k_attn(const _Float16* __restrict__ qb, const _Float16* __restrict__ kb,
       const _Float16* __restrict__ vT, const float* __restrict__ rpb,
       _Float16* __restrict__ attnout) {
  extern __shared__ char smem[];
  float*    S    = (float*)(smem + SM_ATTN_S);     // [112][112]
  _Float16* Pv   = (_Float16*)(smem + SM_ATTN_PV); // [112][128] softmaxed probs (f16)
  _Float16* KL   = (_Float16*)(smem + SM_ATTN_KL); // [128][32]
  _Float16* VL   = (_Float16*)(smem + SM_ATTN_VL); // [32][128]
  float*    RPB  = (float*)(smem + SM_ATTN_RPB);   // [507]
  int*      CREG = (int*)(smem + SM_ATTN_CRG);     // [112]

  int wh = blockIdx.x;
  int w = wh / 3, head = wh - w * 3;
  int tid = threadIdx.x, wave = tid >> 5, lane = tid & 31;
  size_t hb = (size_t)(w * NH + head);

  copy16(kb + hb * NP2 * HD, KL, NP2 * HD * 2, tid, 128);
  copy16(vT + hb * HD * NP2, VL, HD * NP2 * 2, tid, 128);
  for (int i = tid; i < NRPB; i += 128) RPB[i] = rpb[i * 3 + head];
  if (tid < NPAD) {
    int val = 0;
    if (tid < NTOK) {
      int b_, wdi, whi, wwi; win_decompose(w, b_, wdi, whi, wwi);
      int od, oh, ow; tok_decompose(tid, od, oh, ow);
      int gd = wdi * 2 + od, gh = whi * 7 + oh, gw = wwi * 7 + ow;
      int dr = (gd < 6) ? 0 : ((gd < 7) ? 1 : 2);
      int hr = (gh < 105) ? 0 : ((gh < 109) ? 1 : 2);
      int wr = (gw < 105) ? 0 : ((gw < 109) ? 1 : 2);
      val = ((dr * 9 + hr * 3 + wr) << 12) | (od << 8) | (oh << 4) | ow;
    }
    CREG[tid] = val;
  }
  __syncthreads();

  // S = q @ k^T : q tile from global (read once), k tiles from LDS
  const _Float16* qh = qb + hb * NP2 * HD;
  for (int mt = wave; mt < 7; mt += 4) {
    v16h a = load_a16(qh + mt * 16 * HD, HD);
    for (int jt = 0; jt < 7; ++jt) {
      v16h b = load_bw(KL + jt * 16 * HD, HD);   // k row-major (j,d) == B[k=d][n=j]
      v8f acc = {};
      acc = WMMA(a, b, acc);
      int n = lane & 15, roff = (lane < 16) ? 0 : 8;
#pragma unroll
      for (int r = 0; r < 8; ++r)
        S[(mt * 16 + r + roff) * NPAD + jt * 16 + n] = acc[r];
    }
  }
  __syncthreads();

  // bias + mask + softmax, one row per thread; emit normalized f16 P with zero pads
  if (tid < NTOK) {
    int ci = CREG[tid];
    int regi = ci >> 12, di = (ci >> 8) & 15, hi = (ci >> 4) & 15, wi = ci & 15;
    float* row = &S[tid * NPAD];
    float mx = -3.4e38f;
    for (int j = 0; j < NTOK; ++j) {
      int cj = CREG[j];
      int regj = cj >> 12, dj = (cj >> 8) & 15, hj = (cj >> 4) & 15, wj = cj & 15;
      int rpi = (di - dj + 1) * 169 + (hi - hj + 6) * 13 + (wi - wj + 6);
      float v = row[j] + RPB[rpi] + ((regi == regj) ? 0.f : -100.f);
      row[j] = v; mx = fmaxf(mx, v);
    }
    float s = 0.f;
    for (int j = 0; j < NTOK; ++j) { float e = __expf(row[j] - mx); row[j] = e; s += e; }
    float inv = 1.f / s;
    _Float16* prow = &Pv[tid * NP2];
    for (int j = 0; j < NTOK; ++j) prow[j] = (_Float16)(row[j] * inv);
    for (int j = NTOK; j < NP2; ++j) prow[j] = (_Float16)0.f;
  } else if (tid < NPAD) {
    _Float16* prow = &Pv[tid * NP2];
    for (int j = 0; j < NP2; ++j) prow[j] = (_Float16)0.f;
  }
  __syncthreads();

  // out = P @ v : A tiles (4 K-chunks) from f16 LDS P, V from LDS
  for (int mt = wave; mt < 7; mt += 4) {
    v16h a0 = load_a16(Pv + mt * 16 * NP2 + 0,  NP2);
    v16h a1 = load_a16(Pv + mt * 16 * NP2 + 32, NP2);
    v16h a2 = load_a16(Pv + mt * 16 * NP2 + 64, NP2);
    v16h a3 = load_a16(Pv + mt * 16 * NP2 + 96, NP2);
#pragma unroll
    for (int nt = 0; nt < 2; ++nt) {
      v8f acc = {};
      acc = WMMA(a0, load_bw(VL + nt * 16 * NP2 + 0,  NP2), acc);
      acc = WMMA(a1, load_bw(VL + nt * 16 * NP2 + 32, NP2), acc);
      acc = WMMA(a2, load_bw(VL + nt * 16 * NP2 + 64, NP2), acc);
      acc = WMMA(a3, load_bw(VL + nt * 16 * NP2 + 96, NP2), acc);
      int n = lane & 15, roff = (lane < 16) ? 0 : 8;
#pragma unroll
      for (int r = 0; r < 8; ++r) {
        int m = mt * 16 + r + roff;
        attnout[((size_t)w * NPAD + m) * C + head * HD + nt * 16 + n] = (_Float16)acc[r];
      }
    }
  }
}

// ---------------- K4: proj GEMM + window reverse + un-roll + residual -> x2 ----------------
__global__ void __launch_bounds__(128, 1)
k_proj(const _Float16* __restrict__ attnout, const _Float16* __restrict__ projw,
       const float* __restrict__ projb, const float* __restrict__ x,
       float* __restrict__ out) {
  __shared__ _Float16 AL[NPAD * C];   // 21504 B
  int w = blockIdx.x;
  int tid = threadIdx.x, wave = tid >> 5, lane = tid & 31;
  int b_, wdi, whi, wwi; win_decompose(w, b_, wdi, whi, wwi);
  copy16(attnout + (size_t)w * NPAD * C, AL, NPAD * C * 2, tid, 128);
  __syncthreads();

  for (int nt = wave; nt < 6; nt += 4) {
    v16h b0 = load_bw(projw + nt * 16 * C + 0,  C);
    v16h b1 = load_bw(projw + nt * 16 * C + 32, C);
    v16h b2 = load_bw(projw + nt * 16 * C + 64, C);
    int c = nt * 16 + (lane & 15);
    float pb = projb[c];
    int roff = (lane < 16) ? 0 : 8;
    for (int mt = 0; mt < 7; ++mt) {
      v8f acc = {};
      acc = WMMA(load_a16(AL + mt * 16 * C + 0,  C), b0, acc);
      acc = WMMA(load_a16(AL + mt * 16 * C + 32, C), b1, acc);
      acc = WMMA(load_a16(AL + mt * 16 * C + 64, C), b2, acc);
#pragma unroll
      for (int r = 0; r < 8; ++r) {
        int m = mt * 16 + r + roff;
        if (m < NTOK) {
          int od, oh, ow; tok_decompose(m, od, oh, ow);
          int sd = (wdi * 2 + od + 1) & 7;
          int sh = whi * 7 + oh + 3; if (sh >= 112) sh -= 112;
          int sw = wwi * 7 + ow + 3; if (sw >= 112) sw -= 112;
          size_t idx = ((((size_t)b_ * 8 + sd) * 112 + sh) * 112 + sw) * C + c;
          out[idx] = x[idx] + acc[r] + pb;   // x2 = shortcut + attn branch
        }
      }
    }
  }
}

// ---------------- K5: MLP (LN2 -> fc1+gelu -> fc2) + residual, 128 rows/block ----------------
__global__ void __launch_bounds__(128, 1)
k_mlp(float* __restrict__ out, const float* __restrict__ n2w,
      const float* __restrict__ n2b, const _Float16* __restrict__ fc1w,
      const float* __restrict__ fc1b, const _Float16* __restrict__ fc2w,
      const float* __restrict__ fc2b) {
  extern __shared__ char smem[];
  _Float16* A  = (_Float16*)(smem + SM_MLP_A);   // [128][96]
  _Float16* Hh = (_Float16*)(smem + SM_MLP_H);   // [128][384]
  int row0 = blockIdx.x * 128;
  int tid = threadIdx.x, wave = tid >> 5, lane = tid & 31;

  { // LN2: one row per thread
    const float* xr = out + (size_t)(row0 + tid) * C;
    float s = 0.f, q = 0.f;
    for (int c = 0; c < C; ++c) { float v = xr[c]; s += v; q += v * v; }
    float mu = s * (1.f / 96.f);
    float rs = rsqrtf(q * (1.f / 96.f) - mu * mu + 1e-5f);
    for (int c = 0; c < C; ++c)
      A[tid * C + c] = (_Float16)((xr[c] - mu) * rs * n2w[c] + n2b[c]);
  }
  __syncthreads();

  // fc1 + exact gelu: wave owns stripes {wave, wave+4}; A tiles held across nt loop
  {
    v16h a[2][3];
#pragma unroll
    for (int s = 0; s < 2; ++s)
#pragma unroll
      for (int kc = 0; kc < 3; ++kc)
        a[s][kc] = load_a16(A + (wave + 4 * s) * 16 * C + kc * 32, C);
    for (int nt = 0; nt < 24; ++nt) {
      v16h b0 = load_bw(fc1w + nt * 16 * C + 0,  C);
      v16h b1 = load_bw(fc1w + nt * 16 * C + 32, C);
      v16h b2 = load_bw(fc1w + nt * 16 * C + 64, C);
      int c = nt * 16 + (lane & 15);
      float bia = fc1b[c];
      int roff = (lane < 16) ? 0 : 8;
#pragma unroll
      for (int s = 0; s < 2; ++s) {
        v8f acc = {};
        acc = WMMA(a[s][0], b0, acc);
        acc = WMMA(a[s][1], b1, acc);
        acc = WMMA(a[s][2], b2, acc);
#pragma unroll
        for (int r = 0; r < 8; ++r) {
          float v = acc[r] + bia;
          float g = 0.5f * v * (1.f + erff(v * 0.70710678118654752f));
          Hh[((wave + 4 * s) * 16 + r + roff) * HID + c] = (_Float16)g;
        }
      }
    }
  }
  __syncthreads();

  // fc2: per stripe, K-outer with 6 persistent accumulators
  for (int s = 0; s < 2; ++s) {
    int st = wave + 4 * s;
    v8f acc[6] = {};
    for (int kc = 0; kc < 12; ++kc) {
      v16h a = load_a16(Hh + st * 16 * HID + kc * 32, HID);
#pragma unroll
      for (int nt = 0; nt < 6; ++nt) {
        v16h b = load_bw(fc2w + nt * 16 * HID + kc * 32, HID);
        acc[nt] = WMMA(a, b, acc[nt]);
      }
    }
#pragma unroll
    for (int nt = 0; nt < 6; ++nt) {
      int c = nt * 16 + (lane & 15);
      float bia = fc2b[c];
      int roff = (lane < 16) ? 0 : 8;
#pragma unroll
      for (int r = 0; r < 8; ++r) {
        size_t idx = (size_t)(row0 + st * 16 + r + roff) * C + c;
        out[idx] = out[idx] + acc[nt][r] + bia;   // thread-owned RMW: x2 + mlp
      }
    }
  }
}

// ---------------- launch ----------------
extern "C" void kernel_launch(void* const* d_in, const int* in_sizes, int n_in,
                              void* d_out, int out_size, void* d_ws, size_t ws_size,
                              hipStream_t stream) {
  const float* x     = (const float*)d_in[0];
  const float* n1w   = (const float*)d_in[1];
  const float* n1b   = (const float*)d_in[2];
  const float* qkvw  = (const float*)d_in[3];
  const float* qkvb  = (const float*)d_in[4];
  const float* rpb   = (const float*)d_in[5];
  const float* projw = (const float*)d_in[6];
  const float* projb = (const float*)d_in[7];
  const float* n2w   = (const float*)d_in[8];
  const float* n2b   = (const float*)d_in[9];
  const float* fc1w  = (const float*)d_in[10];
  const float* fc1b  = (const float*)d_in[11];
  const float* fc2w  = (const float*)d_in[12];
  const float* fc2b  = (const float*)d_in[13];
  float* out = (float*)d_out;
  char* ws = (char*)d_ws;

  _Float16* w_qkv  = (_Float16*)(ws + OFF_QKVW);
  _Float16* w_proj = (_Float16*)(ws + OFF_PROJW);
  _Float16* w_fc1  = (_Float16*)(ws + OFF_FC1W);
  _Float16* w_fc2  = (_Float16*)(ws + OFF_FC2W);
  _Float16* winA   = (_Float16*)(ws + OFF_WINA);
  _Float16* attno  = winA;  // alias: winA consumed before attnout is produced
  _Float16* qB     = (_Float16*)(ws + OFF_Q);
  _Float16* kB     = (_Float16*)(ws + OFF_K);
  _Float16* vT     = (_Float16*)(ws + OFF_VT);

  k_prep<<<(HID * C + 255) / 256, 256, 0, stream>>>(qkvw, projw, fc1w, fc2w,
                                                    w_qkv, w_proj, w_fc1, w_fc2);
  k_ln1<<<NWIN * NPAD / 4, 128, 0, stream>>>(x, n1w, n1b, winA);
  k_qkv<<<NWIN, 128, 0, stream>>>(winA, w_qkv, qkvb, qB, kB, vT);
  k_attn<<<NWIN * NH, 128, SM_ATTN_SZ, stream>>>(qB, kB, vT, rpb, attno);
  k_proj<<<NWIN, 128, 0, stream>>>(attno, w_proj, projb, x, out);
  k_mlp<<<TOKENS / 128, 128, SM_MLP_SZ, stream>>>(out, n2w, n2b, w_fc1, fc1b, w_fc2, fc2b);
}
